// SwinTransformerBlock_76647986364434
// MI455X (gfx1250) — matrix operand order repaired and checked by reading, not compiled, not run on hardware
//
#include <hip/hip_runtime.h>
#include <hip/hip_bf16.h>

// ---------------------------------------------------------------------------
// Swin block on gfx1250 (MI455X): all GEMMs via v_wmma_f32_16x16x32_bf16
// (wave32). bf16 activations between GEMMs (halves HBM traffic), f32
// accumulate. Big GEMMs use 32x64-per-wave blocking: 2 A-frags + 4 B-frags
// feed 8 WMMAs per k-step (1.5 b128 loads per WMMA vs 4 with 16x16 tiles).
// ---------------------------------------------------------------------------

typedef __bf16 bf16_t;
typedef __attribute__((ext_vector_type(16))) __bf16 v16bf;
typedef __attribute__((ext_vector_type(8)))  __bf16 v8bf;
typedef __attribute__((ext_vector_type(8)))  float  v8f;

#define B_    32
#define H_    56
#define W_    56
#define C_    384
#define WS_   8
#define SH_   4
#define HEADS_ 8
#define HD_   48
#define NWH   7
#define NW_PER_B 49
#define NWIN  1568      /* B_*49 */
#define NTOK  100352    /* B_*H_*W_ */
#define NEGBIG (-1e9f)

// ---- bf16 <-> f32 (RNE), bit-level to avoid relying on __bf16 arithmetic ----
static __device__ __forceinline__ bf16_t f2bf(float f) {
  union { float f; unsigned u; } v; v.f = f;
  unsigned r = v.u + 0x7FFFu + ((v.u >> 16) & 1u);
  union { unsigned short s; bf16_t b; } o; o.s = (unsigned short)(r >> 16);
  return o.b;
}

static __device__ __forceinline__ v8f zero8() {
  v8f z;
#pragma unroll
  for (int i = 0; i < 8; ++i) z[i] = 0.f;
  return z;
}

static __device__ __forceinline__ v8f wmma_bf16(v16bf a, v16bf b, v8f c) {
  return __builtin_amdgcn_wmma_f32_16x16x32_bf16(false, a, false, b, (short)0, c,
                                                 false, false);
}

// A fragment: 16x32 tile, row-major, stride lda (elements).
// Contiguous k order, matching load_b_frag: lane<16 holds row=lane, K 0-15; lane>=16 row=lane-16, K 16-31.
static __device__ __forceinline__ v16bf load_a_frag(const bf16_t* A, int lda) {
  int lane = threadIdx.x & 31;
  const bf16_t* p = A + (lane & 15) * lda + ((lane >> 4) << 4);
  v8bf lo = *(const v8bf*)(p);
  v8bf hi = *(const v8bf*)(p + 8);
  return __builtin_shufflevector(lo, hi, 0,1,2,3,4,5,6,7,8,9,10,11,12,13,14,15);
}

// B fragment from B-transposed storage Bt[Ncols][K] (each lane reads one
// column of B contiguously). lane<16: col=lane, K 0-15; lane>=16: col=lane-16, K 16-31.
static __device__ __forceinline__ v16bf load_b_frag(const bf16_t* Bt, int ldb) {
  int lane = threadIdx.x & 31;
  return *(const v16bf*)(Bt + (lane & 15) * ldb + ((lane >> 4) << 4));
}

// 32x64 output block per wave: A[32 x K] row-major, Bt[64 x K] row-major.
// lda == ldb == K for all our GEMMs.
static __device__ __forceinline__ void gemm_wave_32x64(
    const bf16_t* __restrict__ Ap, const bf16_t* __restrict__ Bp, int K,
    v8f acc[2][4]) {
  for (int k0 = 0; k0 < K; k0 += 32) {
    __builtin_prefetch(Ap + k0 + 128, 0, 3);
    v16bf a0 = load_a_frag(Ap + k0, K);
    v16bf a1 = load_a_frag(Ap + 16 * K + k0, K);
#pragma unroll
    for (int u = 0; u < 4; ++u) {
      v16bf b = load_b_frag(Bp + (u * 16) * K + k0, K);
      acc[0][u] = wmma_bf16(a0, b, acc[0][u]);
      acc[1][u] = wmma_bf16(a1, b, acc[1][u]);
    }
  }
}

// ---------------------------------------------------------------------------
// K0: f32 weight [R][Ccols] -> bf16 transposed [Ccols][R]
__global__ void wt_transpose_kernel(const float* __restrict__ src,
                                    bf16_t* __restrict__ dst, int R, int Ccols) {
  int idx = blockIdx.x * blockDim.x + threadIdx.x;
  if (idx >= R * Ccols) return;
  int i = idx / Ccols, j = idx % Ccols;
  dst[j * R + i] = f2bf(src[idx]);
}

// ---------------------------------------------------------------------------
// K1: LayerNorm1 + roll(-4,-4) + window partition -> winX bf16 [NTOK][C]
__global__ void ln1_window_kernel(const float* __restrict__ patches,
                                  const float* __restrict__ w,
                                  const float* __restrict__ bvec,
                                  bf16_t* __restrict__ winX) {
  int t = blockIdx.x;
  int win_g = t >> 6, n = t & 63;
  int bb = win_g / NW_PER_B, wwin = win_g % NW_PER_B;
  int wh = wwin / NWH, ww = wwin % NWH;
  int r = n >> 3, c2 = n & 7;
  int sh = (wh * 8 + r + SH_) % H_;
  int sw = (ww * 8 + c2 + SH_) % W_;
  const float* src = patches + ((size_t)(bb * H_ + sh) * W_ + sw) * C_;
  int tid = threadIdx.x;
  __shared__ float rs[128], rq[128];
  float x0 = src[tid], x1 = src[tid + 128], x2 = src[tid + 256];
  rs[tid] = x0 + x1 + x2;
  rq[tid] = x0 * x0 + x1 * x1 + x2 * x2;
  __syncthreads();
  for (int o = 64; o > 0; o >>= 1) {
    if (tid < o) { rs[tid] += rs[tid + o]; rq[tid] += rq[tid + o]; }
    __syncthreads();
  }
  float mean = rs[0] * (1.f / C_);
  float var  = rq[0] * (1.f / C_) - mean * mean;
  float rstd = rsqrtf(var + 1e-5f);
  bf16_t* dst = winX + (size_t)t * C_;
  dst[tid]       = f2bf((x0 - mean) * rstd * w[tid]       + bvec[tid]);
  dst[tid + 128] = f2bf((x1 - mean) * rstd * w[tid + 128] + bvec[tid + 128]);
  dst[tid + 256] = f2bf((x2 - mean) * rstd * w[tid + 256] + bvec[tid + 256]);
}

// ---------------------------------------------------------------------------
// K2: QKV GEMM [NTOK,384]x[384,1152]; 32x64 per wave; scatter:
// Q (scaled) / K row-major [win][h][n][48], V transposed [win][h][48][n].
__global__ __launch_bounds__(256) void qkv_gemm_kernel(
    const bf16_t* __restrict__ X, const bf16_t* __restrict__ Wt,
    const float* __restrict__ bias, bf16_t* __restrict__ qbuf,
    bf16_t* __restrict__ kbuf, bf16_t* __restrict__ vtbuf) {
  int wave = (blockIdx.x * 256 + threadIdx.x) >> 5;
  const int NG = (3 * C_) / 64;  // 18 col groups of 64
  int mt = wave / NG, ng = wave % NG;
  int m0 = mt * 32, n0 = ng * 64;
  const bf16_t* Ap = X + (size_t)m0 * C_;
  const bf16_t* Bp = Wt + (size_t)n0 * C_;
  v8f acc[2][4];
#pragma unroll
  for (int mi = 0; mi < 2; ++mi)
#pragma unroll
    for (int u = 0; u < 4; ++u) acc[mi][u] = zero8();
  gemm_wave_32x64(Ap, Bp, C_, acc);

  int lane = threadIdx.x & 31;
  int rowbase = (lane >> 4) << 3;
#pragma unroll
  for (int u = 0; u < 4; ++u) {
    int j = n0 + u * 16 + (lane & 15);
    int s = j / C_;
    int h = (j % C_) / HD_;
    int d = j % HD_;
    float bv = bias[j];
    float scale = (s == 0) ? 0.14433756729740643f : 1.f;  // 1/sqrt(48)
#pragma unroll
    for (int mi = 0; mi < 2; ++mi) {
#pragma unroll
      for (int r = 0; r < 8; ++r) {
        int t = m0 + mi * 16 + rowbase + r;
        int win = t >> 6, n = t & 63;
        float val = (acc[mi][u][r] + bv) * scale;
        if (s == 0)
          qbuf[(size_t)((win * HEADS_ + h) * 64 + n) * HD_ + d] = f2bf(val);
        else if (s == 1)
          kbuf[(size_t)((win * HEADS_ + h) * 64 + n) * HD_ + d] = f2bf(val);
        else
          vtbuf[(size_t)((win * HEADS_ + h) * HD_ + d) * 64 + n] = f2bf(val);
      }
    }
  }
}

// ---------------------------------------------------------------------------
// K3: attention per (window, head). QK^T + rel-bias + shift-mask, softmax, PV.
__global__ __launch_bounds__(256) void attn_kernel(
    const bf16_t* __restrict__ qbuf, const bf16_t* __restrict__ kbuf,
    const bf16_t* __restrict__ vtbuf, const float* __restrict__ relTab,
    bf16_t* __restrict__ attnout) {
  __shared__ bf16_t sQ[64][80];   // head-dim padded 48 -> 64 (cols 64..79 pad)
  __shared__ bf16_t sK[64][80];
  __shared__ float  sS[64][80];
  __shared__ bf16_t sP[64][80];

  int wg = blockIdx.x;
  int win = wg >> 3, h = wg & 7;
  int wwin = win % NW_PER_B;
  int wh = wwin / NWH, ww = wwin % NWH;
  int tid = threadIdx.x;
  int wv = tid >> 5, lane = tid & 31;
  bf16_t bz = f2bf(0.f);

  size_t base = (size_t)(win * HEADS_ + h) * 64 * HD_;
  for (int i = tid; i < 64 * 64; i += 256) {
    int row = i >> 6, col = i & 63;
    sQ[row][col] = (col < HD_) ? qbuf[base + row * HD_ + col] : bz;
    sK[row][col] = (col < HD_) ? kbuf[base + row * HD_ + col] : bz;
  }
  __syncthreads();

  // S = Q K^T ; 16 tiles of 16x16, 2 per wave
  for (int tile = wv; tile < 16; tile += 8) {
    int ti = tile >> 2, tj = tile & 3;
    int m0 = ti * 16, n0 = tj * 16;
    v8f acc = zero8();
    acc = wmma_bf16(load_a_frag(&sQ[m0][0], 80),  load_b_frag(&sK[n0][0], 80),  acc);
    acc = wmma_bf16(load_a_frag(&sQ[m0][32], 80), load_b_frag(&sK[n0][32], 80), acc);
    int mcol = n0 + (lane & 15);
    int r2 = mcol >> 3, c2k = mcol & 7;
    int gr2 = wh * 8 + r2, gc2 = ww * 8 + c2k;
    int lab2 = (gr2 < 48 ? 0 : (gr2 < 52 ? 1 : 2)) * 3 +
               (gc2 < 48 ? 0 : (gc2 < 52 ? 1 : 2));
    int rowbase = (lane >> 4) << 3;
#pragma unroll
    for (int r = 0; r < 8; ++r) {
      int nrow = m0 + rowbase + r;
      int r1 = nrow >> 3, c1 = nrow & 7;
      int gr1 = wh * 8 + r1, gc1 = ww * 8 + c1;
      int lab1 = (gr1 < 48 ? 0 : (gr1 < 52 ? 1 : 2)) * 3 +
                 (gc1 < 48 ? 0 : (gc1 < 52 ? 1 : 2));
      int ridx = (r1 - r2 + 7) * 15 + (c1 - c2k + 7);
      float bias = relTab[ridx * HEADS_ + h];
      float msk  = (lab1 != lab2) ? NEGBIG : 0.f;
      sS[nrow][mcol] = acc[r] + bias + msk;
    }
  }
  __syncthreads();

  // softmax: wave wv handles rows 8*wv .. 8*wv+7; 2 cols per lane (wave32)
  for (int row = wv * 8; row < wv * 8 + 8; ++row) {
    float v0 = sS[row][lane], v1 = sS[row][lane + 32];
    float mx = fmaxf(v0, v1);
    for (int o = 16; o; o >>= 1) mx = fmaxf(mx, __shfl_xor(mx, o, 32));
    float e0 = __expf(v0 - mx), e1 = __expf(v1 - mx);
    float sm = e0 + e1;
    for (int o = 16; o; o >>= 1) sm += __shfl_xor(sm, o, 32);
    float inv = 1.f / sm;
    sP[row][lane]      = f2bf(e0 * inv);
    sP[row][lane + 32] = f2bf(e1 * inv);
  }
  __syncthreads();

  // O = P V : [64x64] @ [64x48]; 12 tiles, V already transposed in vtbuf
  const bf16_t* Vt = vtbuf + (size_t)(win * HEADS_ + h) * HD_ * 64;
  for (int tile = wv; tile < 12; tile += 8) {
    int ti = tile / 3, tj = tile % 3;
    int m0 = ti * 16, d0 = tj * 16;
    v8f acc = zero8();
    acc = wmma_bf16(load_a_frag(&sP[m0][0], 80),  load_b_frag(Vt + d0 * 64, 64),      acc);
    acc = wmma_bf16(load_a_frag(&sP[m0][32], 80), load_b_frag(Vt + d0 * 64 + 32, 64), acc);
    int dd = d0 + (lane & 15);
    int rowbase = (lane >> 4) << 3;
#pragma unroll
    for (int r = 0; r < 8; ++r) {
      int nrow = m0 + rowbase + r;
      attnout[(size_t)(win * 64 + nrow) * C_ + h * HD_ + dd] = f2bf(acc[r]);
    }
  }
}

// ---------------------------------------------------------------------------
// K4: proj GEMM + reverse window partition + roll(+4,+4) + residual -> xbuf f32
__global__ __launch_bounds__(256) void proj_gemm_kernel(
    const bf16_t* __restrict__ A, const bf16_t* __restrict__ Wt,
    const float* __restrict__ bias, const float* __restrict__ patches,
    float* __restrict__ xbuf) {
  int wave = (blockIdx.x * 256 + threadIdx.x) >> 5;
  const int NG = C_ / 64;  // 6
  int mt = wave / NG, ng = wave % NG;
  int m0 = mt * 32, n0 = ng * 64;
  v8f acc[2][4];
#pragma unroll
  for (int mi = 0; mi < 2; ++mi)
#pragma unroll
    for (int u = 0; u < 4; ++u) acc[mi][u] = zero8();
  gemm_wave_32x64(A + (size_t)m0 * C_, Wt + (size_t)n0 * C_, C_, acc);

  int lane = threadIdx.x & 31;
  int rowbase = (lane >> 4) << 3;
#pragma unroll
  for (int u = 0; u < 4; ++u) {
    int c = n0 + u * 16 + (lane & 15);
    float bv = bias[c];
#pragma unroll
    for (int mi = 0; mi < 2; ++mi) {
#pragma unroll
      for (int r = 0; r < 8; ++r) {
        int t = m0 + mi * 16 + rowbase + r;
        int win_g = t >> 6, n = t & 63;
        int bb = win_g / NW_PER_B, wwin = win_g % NW_PER_B;
        int wh = wwin / NWH, ww = wwin % NWH;
        int rr = n >> 3, cc = n & 7;
        int oh = (wh * 8 + rr + SH_) % H_;
        int ow = (ww * 8 + cc + SH_) % W_;
        size_t pix = ((size_t)(bb * H_ + oh) * W_ + ow) * C_ + c;
        xbuf[pix] = patches[pix] + acc[mi][u][r] + bv;
      }
    }
  }
}

// ---------------------------------------------------------------------------
// K5: LayerNorm2 -> bf16
__global__ void ln2_kernel(const float* __restrict__ xbuf,
                           const float* __restrict__ w,
                           const float* __restrict__ bvec,
                           bf16_t* __restrict__ out) {
  int t = blockIdx.x, tid = threadIdx.x;
  const float* src = xbuf + (size_t)t * C_;
  __shared__ float rs[128], rq[128];
  float x0 = src[tid], x1 = src[tid + 128], x2 = src[tid + 256];
  rs[tid] = x0 + x1 + x2;
  rq[tid] = x0 * x0 + x1 * x1 + x2 * x2;
  __syncthreads();
  for (int o = 64; o > 0; o >>= 1) {
    if (tid < o) { rs[tid] += rs[tid + o]; rq[tid] += rq[tid + o]; }
    __syncthreads();
  }
  float mean = rs[0] * (1.f / C_);
  float var  = rq[0] * (1.f / C_) - mean * mean;
  float rstd = rsqrtf(var + 1e-5f);
  bf16_t* dst = out + (size_t)t * C_;
  dst[tid]       = f2bf((x0 - mean) * rstd * w[tid]       + bvec[tid]);
  dst[tid + 128] = f2bf((x1 - mean) * rstd * w[tid + 128] + bvec[tid + 128]);
  dst[tid + 256] = f2bf((x2 - mean) * rstd * w[tid + 256] + bvec[tid + 256]);
}

// ---------------------------------------------------------------------------
// K6: fc1 GEMM + GELU(tanh) -> h1 bf16 [NTOK][1536]
__global__ __launch_bounds__(256) void fc1_gemm_kernel(
    const bf16_t* __restrict__ A, const bf16_t* __restrict__ Wt,
    const float* __restrict__ bias, bf16_t* __restrict__ out) {
  int wave = (blockIdx.x * 256 + threadIdx.x) >> 5;
  const int NG = (4 * C_) / 64;  // 24
  int mt = wave / NG, ng = wave % NG;
  int m0 = mt * 32, n0 = ng * 64;
  v8f acc[2][4];
#pragma unroll
  for (int mi = 0; mi < 2; ++mi)
#pragma unroll
    for (int u = 0; u < 4; ++u) acc[mi][u] = zero8();
  gemm_wave_32x64(A + (size_t)m0 * C_, Wt + (size_t)n0 * C_, C_, acc);

  int lane = threadIdx.x & 31;
  int rowbase = (lane >> 4) << 3;
#pragma unroll
  for (int u = 0; u < 4; ++u) {
    int c = n0 + u * 16 + (lane & 15);
    float bv = bias[c];
#pragma unroll
    for (int mi = 0; mi < 2; ++mi) {
#pragma unroll
      for (int r = 0; r < 8; ++r) {
        int t = m0 + mi * 16 + rowbase + r;
        float x = acc[mi][u][r] + bv;
        float g = 0.5f * x * (1.f + tanhf(0.7978845608f * (x + 0.044715f * x * x * x)));
        out[(size_t)t * (4 * C_) + c] = f2bf(g);
      }
    }
  }
}

// ---------------------------------------------------------------------------
// K7: fc2 GEMM + residual -> d_out f32
__global__ __launch_bounds__(256) void fc2_gemm_kernel(
    const bf16_t* __restrict__ A, const bf16_t* __restrict__ Wt,
    const float* __restrict__ bias, const float* __restrict__ xbuf,
    float* __restrict__ out) {
  int wave = (blockIdx.x * 256 + threadIdx.x) >> 5;
  const int NG = C_ / 64;  // 6
  int mt = wave / NG, ng = wave % NG;
  int m0 = mt * 32, n0 = ng * 64;
  const int K4 = 4 * C_;
  v8f acc[2][4];
#pragma unroll
  for (int mi = 0; mi < 2; ++mi)
#pragma unroll
    for (int u = 0; u < 4; ++u) acc[mi][u] = zero8();
  gemm_wave_32x64(A + (size_t)m0 * K4, Wt + (size_t)n0 * K4, K4, acc);

  int lane = threadIdx.x & 31;
  int rowbase = (lane >> 4) << 3;
#pragma unroll
  for (int u = 0; u < 4; ++u) {
    int c = n0 + u * 16 + (lane & 15);
    float bv = bias[c];
#pragma unroll
    for (int mi = 0; mi < 2; ++mi) {
#pragma unroll
      for (int r = 0; r < 8; ++r) {
        int t = m0 + mi * 16 + rowbase + r;
        size_t idx = (size_t)t * C_ + c;
        out[idx] = xbuf[idx] + acc[mi][u][r] + bv;
      }
    }
  }
}

// ---------------------------------------------------------------------------
extern "C" void kernel_launch(void* const* d_in, const int* in_sizes, int n_in,
                              void* d_out, int out_size, void* d_ws, size_t ws_size,
                              hipStream_t stream) {
  (void)in_sizes; (void)n_in; (void)out_size; (void)ws_size;
  const float* patches = (const float*)d_in[0];
  const float* ln1_w   = (const float*)d_in[1];
  const float* ln1_b   = (const float*)d_in[2];
  const float* qkv_w   = (const float*)d_in[3];
  const float* qkv_b   = (const float*)d_in[4];
  const float* proj_w  = (const float*)d_in[5];
  const float* proj_b  = (const float*)d_in[6];
  const float* relTab  = (const float*)d_in[7];
  const float* ln2_w   = (const float*)d_in[8];
  const float* ln2_b   = (const float*)d_in[9];
  const float* fc1_w   = (const float*)d_in[10];
  const float* fc1_b   = (const float*)d_in[11];
  const float* fc2_w   = (const float*)d_in[12];
  const float* fc2_b   = (const float*)d_in[13];

  char* ws = (char*)d_ws;
  bf16_t* qkv_wt  = (bf16_t*)(ws + 0);            //  884736 B
  bf16_t* proj_wt = (bf16_t*)(ws + 884736);       //  294912 B
  bf16_t* fc1_wt  = (bf16_t*)(ws + 1179648);      // 1179648 B
  bf16_t* fc2_wt  = (bf16_t*)(ws + 2359296);      // 1179648 B
  bf16_t* winX    = (bf16_t*)(ws + 3538944);      // 77070336 B
  bf16_t* qbuf    = (bf16_t*)(ws + 80609280);     // 77070336 B
  bf16_t* kbuf    = (bf16_t*)(ws + 157679616);    // 77070336 B
  bf16_t* vtbuf   = (bf16_t*)(ws + 234749952);    // 77070336 B
  bf16_t* attnout = (bf16_t*)(ws + 311820288);    // 77070336 B
  float*  xbuf    = (float*) (ws + 388890624);    // 154140672 B
  bf16_t* ln2buf  = (bf16_t*)(ws + 543031296);    // 77070336 B
  bf16_t* h1      = (bf16_t*)(ws + 620101632);    // 308281344 B  (total ~928 MB)

  // weight prep (bf16, B-transposed)
  wt_transpose_kernel<<<(384 * 1152 + 255) / 256, 256, 0, stream>>>(qkv_w, qkv_wt, 384, 1152);
  wt_transpose_kernel<<<(384 * 384  + 255) / 256, 256, 0, stream>>>(proj_w, proj_wt, 384, 384);
  wt_transpose_kernel<<<(384 * 1536 + 255) / 256, 256, 0, stream>>>(fc1_w, fc1_wt, 384, 1536);
  wt_transpose_kernel<<<(1536 * 384 + 255) / 256, 256, 0, stream>>>(fc2_w, fc2_wt, 1536, 384);

  // LN1 + shift + window partition
  ln1_window_kernel<<<NTOK, 128, 0, stream>>>(patches, ln1_w, ln1_b, winX);

  // QKV GEMM: 3136 row-tiles(32) x 18 col-groups(64) = 56448 waves / 8
  qkv_gemm_kernel<<<7056, 256, 0, stream>>>(winX, qkv_wt, qkv_b, qbuf, kbuf, vtbuf);

  // attention: one workgroup per (window, head)
  attn_kernel<<<NWIN * HEADS_, 256, 0, stream>>>(qbuf, kbuf, vtbuf, relTab, attnout);

  // proj + reverse-shift + residual: 3136 x 6 groups / 8
  proj_gemm_kernel<<<2352, 256, 0, stream>>>(attnout, proj_wt, proj_b, patches, xbuf);

  // LN2
  ln2_kernel<<<NTOK, 128, 0, stream>>>(xbuf, ln2_w, ln2_b, ln2buf);

  // fc1 + gelu: 3136 x 24 groups / 8
  fc1_gemm_kernel<<<9408, 256, 0, stream>>>(ln2buf, fc1_wt, fc1_b, h1);

  // fc2 + residual: 3136 x 6 groups / 8
  fc2_gemm_kernel<<<2352, 256, 0, stream>>>(h1, fc2_wt, fc2_b, xbuf, (float*)d_out);
}
